// RSSMCore_65592740544943
// MI455X (gfx1250) — matrix operand order repaired and checked by reading, not compiled, not run on hardware
//
#include <hip/hip_runtime.h>
#include <hip/hip_bf16.h>

// ---------------------------------------------------------------------------
// RSSM core for MI455X (gfx1250): persistent per-batch-tile workgroups,
// bf16 WMMA (16x16x32, f32 accumulate) for all GEMMs, fp32 recurrent state
// in LDS. 16 workgroups x 512 threads (16 wave32s). Weights packed to bf16
// once per launch into d_ws (L2-resident thereafter).
// Round-4 changes: (a) launder an integer zero-offset (volatile asm, depends
// on t) instead of the weight pointers -- keeps kernarg provenance so weight
// fragments use global_load_b128 (SGPR base) instead of flat_load_b128 with
// 64-bit VGPR address math, while still blocking the LICM hoist+spill;
// (b) float4-vectorized embedding staging (b128 loads, merged b64 DS stores).
// ---------------------------------------------------------------------------

typedef __attribute__((ext_vector_type(16))) __bf16 v16bf;
typedef __attribute__((ext_vector_type(8)))  __bf16 v8bf;
typedef __attribute__((ext_vector_type(8)))  float  v8f;

#define A_DIM 6
#define Z_DIM 32
#define H_DIM 512
#define E_DIM 1024
#define TLEN  64

// workspace offsets (bf16 elements)
#define OFF_WAZ 0                         // [512][64]   ([z|a] pad 38->64)
#define OFF_WIH (OFF_WAZ + 512*64)        // [1536][512]
#define OFF_WHH (OFF_WIH + 1536*512)      // [1536][512]
#define OFF_WHA (OFF_WHH + 1536*512)      // [512][544]  ([h|a] pad 518->544)
#define OFF_WPR (OFF_WHA + 512*544)       // [64][512]
#define OFF_WHE (OFF_WPR + 64*512)        // [512][1536]
#define OFF_WPO (OFF_WHE + 512*1536)      // [64][512]

__device__ __forceinline__ __bf16 f2bf(float f) {
  return (__bf16)f;   // native f32->bf16 (v_cvt_pk_bf16_f32)
}

// Per-iteration opaque zero: volatile + consumes t, so it can never be
// hoisted, CSE'd across iterations, or folded. Added to kernarg pointers it
// blocks LICM of weight loads while preserving global-addrspace provenance.
__device__ __forceinline__ int opaque_zero(int t) {
  int x = 0;
  asm volatile("" : "+s"(x) : "s"(t));
  return x;
}

__device__ __forceinline__ float sigmoidf_(float x) {
  return 1.f / (1.f + __expf(-x));
}
__device__ __forceinline__ float softplusf_(float x) {
  if (x > 20.f)  return x;
  if (x < -20.f) return __expf(x);
  return log1pf(__expf(x));
}

// A fragment: 16x32 bf16, rows = M (lane&15). Per ISA 16-bit A layout:
// lane<16 holds K = kb+{0..7,16..23}, lane>=16 holds K = kb+{8..15,24..31}.
__device__ __forceinline__ v16bf ld_afrag(const __bf16* base, int ld, int kb, int lane) {
  const int m  = lane & 15;
  const int hi = lane >> 4;
  const __bf16* p = base + m * ld + kb + hi * 8;
  v8bf lo = *(const v8bf*)(p);        // K = kb + hi*8 + 0..7
  v8bf hh = *(const v8bf*)(p + 16);   // K = kb + 16 + hi*8 + 0..7
  return __builtin_shufflevector(lo, hh, 0,1,2,3,4,5,6,7,8,9,10,11,12,13,14,15);
}

// B fragment: 32x16 bf16 built from row-major weights W[N][K]:
// N = lane&15, lane<16 holds K = kb+{0..7,16..23}, lane>=16 holds K = kb+{8..15,24..31}.
__device__ __forceinline__ v16bf ld_bfrag(const __bf16* __restrict__ base, int ld,
                                          int nbase, int kb, int lane) {
  const int n  = lane & 15;
  const int hi = lane >> 4;
  const __bf16* p = base + (size_t)(nbase + n) * ld + kb + hi * 8;
  v8bf lo = *(const v8bf*)(p);
  v8bf hh = *(const v8bf*)(p + 16);
  return __builtin_shufflevector(lo, hh, 0,1,2,3,4,5,6,7,8,9,10,11,12,13,14,15);
}

__device__ __forceinline__ v8f wmma_bf16(v16bf a, v16bf b, v8f c) {
  return __builtin_amdgcn_wmma_f32_16x16x32_bf16(false, a, false, b, (short)0, c,
                                                 false, false);
}

// ---------------------------------------------------------------------------
// Weight pack: fp32 [rows][scols] -> bf16 [rows][dcols] zero-padded.
// ---------------------------------------------------------------------------
__global__ void pack_bf16(const float* __restrict__ src, __bf16* __restrict__ dst,
                          int rows, int scols, int dcols) {
  int i = blockIdx.x * blockDim.x + threadIdx.x;
  int total = rows * dcols;
  if (i >= total) return;
  int r = i / dcols, c = i - r * dcols;
  float v = (c < scols) ? src[(size_t)r * scols + c] : 0.f;
  dst[i] = f2bf(v);
}

// ---------------------------------------------------------------------------
// Main persistent kernel: blockIdx.x = batch tile (16 rows), 512 threads.
// ---------------------------------------------------------------------------
__global__ void __launch_bounds__(512, 1)
rssm_core(const float* __restrict__ prev_z,
          const float* __restrict__ prev_h,
          const float* __restrict__ actions,
          const float* __restrict__ embeddings,
          const float* __restrict__ dones,
          const float* __restrict__ post_noise,
          const __bf16* __restrict__ wbf,
          const float* __restrict__ b_az,
          const float* __restrict__ b_ih,
          const float* __restrict__ b_hh,
          const float* __restrict__ b_ha,
          const float* __restrict__ b_prior,
          const float* __restrict__ b_he,
          const float* __restrict__ b_post,
          float* __restrict__ out) {
  const int tid  = threadIdx.x;
  const int lane = tid & 31;
  const int wave = tid >> 5;
  const int b0   = blockIdx.x * 16;

  // output tuple, flat: h[B,T,512], z[B,T,32], pm, ps, qm, qs (each [B,T,32])
  const size_t OH = (size_t)256 * TLEN * H_DIM;
  const size_t OZ = (size_t)256 * TLEN * Z_DIM;
  float* out_h  = out;
  float* out_z  = out_h  + OH;
  float* out_pm = out_z  + OZ;
  float* out_ps = out_pm + OZ;
  float* out_qm = out_ps + OZ;
  float* out_qs = out_qm + OZ;

  extern __shared__ unsigned char smem_raw[];
  float*  h_s  = (float*)smem_raw;            // [16][512] fp32 carry
  float*  z_s  = h_s  + 16 * 512;             // [16][32]
  float*  eq_s = z_s  + 16 * 32;              // [16][32] posterior noise
  float*  qm_s = eq_s + 16 * 32;              // [16][32]
  float*  qs_s = qm_s + 16 * 32;              // [16][32]
  __bf16* hbf  = (__bf16*)(qs_s + 16 * 32);   // [16][544]  [h | a | pad]
  __bf16* ebf  = hbf  + 16 * 544;             // [16][1024] embeddings
  __bf16* xbf  = ebf  + 16 * 1024;            // [16][512]  GRU input x
  __bf16* zabf = xbf  + 16 * 512;             // [16][64]   [z | a | pad]
  __bf16* habf = zabf + 16 * 64;              // [16][512]  relu(ha)
  __bf16* hebf = habf + 16 * 512;             // [16][512]  relu(he)

  // ---- init carry ----
  for (int i = tid; i < 16 * H_DIM; i += 512) {
    int r = i >> 9, c = i & 511;
    float v = prev_h[(size_t)(b0 + r) * H_DIM + c];
    h_s[r * 512 + c] = v;
    hbf[r * 544 + c] = f2bf(v);
  }
  for (int i = tid; i < 16 * Z_DIM; i += 512) {
    int r = i >> 5, c = i & 31;
    z_s[r * 32 + c] = prev_z[(size_t)(b0 + r) * Z_DIM + c];
  }
  const __bf16 zbf0 = f2bf(0.f);
  for (int i = tid; i < 16 * 26; i += 512) {
    int r = i / 26, c = i - r * 26;
    hbf[r * 544 + 518 + c] = zbf0;   // pad of [h|a] concat
    zabf[r * 64 + 38 + c]  = zbf0;   // pad of [z|a] concat
  }
  __syncthreads();

  for (int t = 0; t < TLEN; ++t) {
    // Fresh unknown-but-zero offset each step: blocks hoisting of weight-
    // fragment loads out of the time loop (L2 re-fetch beats scratch spill).
    const int zo = opaque_zero(t);
    const __bf16* Waz = wbf + OFF_WAZ + zo;
    const __bf16* Wih = wbf + OFF_WIH + zo;
    const __bf16* Whh = wbf + OFF_WHH + zo;
    const __bf16* Wha = wbf + OFF_WHA + zo;
    const __bf16* Wpr = wbf + OFF_WPR + zo;
    const __bf16* Whe = wbf + OFF_WHE + zo;
    const __bf16* Wpo = wbf + OFF_WPO + zo;

    // ---- stage0: per-step inputs; z *= (1-done) ----
    {
      int r = tid >> 5, c = lane;                 // 16 x 32 = 512 threads
      float d  = dones[(size_t)(b0 + r) * TLEN + t];
      float zv = z_s[r * 32 + c] * (1.f - d);
      z_s[r * 32 + c]  = zv;
      zabf[r * 64 + c] = f2bf(zv);
      eq_s[r * 32 + c] = post_noise[((size_t)(b0 + r) * TLEN + t) * Z_DIM + c];
      if (c < A_DIM) {
        float av = actions[((size_t)(b0 + r) * TLEN + t) * A_DIM + c];
        __bf16 ab = f2bf(av);
        zabf[r * 64 + 32 + c]  = ab;
        hbf[r * 544 + 512 + c] = ab;
      }
    }
    // embeddings: float4 loads (rows 4KB aligned), 4 adjacent bf16 stores
    // merge into ds_store_b64.
    for (int i = tid; i < 16 * (E_DIM / 4); i += 512) {
      int r = i >> 8, c4 = (i & 255) * 4;
      const float4 e4 = *(const float4*)(embeddings +
          ((size_t)(b0 + r) * TLEN + t) * E_DIM + c4);
      __bf16* dst = ebf + r * 1024 + c4;
      dst[0] = f2bf(e4.x); dst[1] = f2bf(e4.y);
      dst[2] = f2bf(e4.z); dst[3] = f2bf(e4.w);
    }
    __syncthreads();

    // ---- stage1: x = relu([z,a] @ Waz^T + b_az)   (K=64, N=512) ----
    for (int nn = 0; nn < 2; ++nn) {
      const int n0 = wave * 32 + nn * 16;
      v8f acc = {};
#pragma unroll
      for (int kt = 0; kt < 2; ++kt) {
        v16bf af = ld_afrag(zabf, 64, kt * 32, lane);
        acc = wmma_bf16(af, ld_bfrag(Waz, 64, n0, kt * 32, lane), acc);
      }
      const int mb = (lane >> 4) * 8;
      const int col = n0 + (lane & 15);
      const float bb = b_az[col];
#pragma unroll
      for (int v = 0; v < 8; ++v)
        xbf[(mb + v) * 512 + col] = f2bf(fmaxf(acc[v] + bb, 0.f));
    }
    __syncthreads();

    // ---- stage2: GRU. Wave w owns h columns [32w, 32w+32).
    //      Two passes per N-tile: pass A -> r,u gates (4 accs),
    //      pass B -> n gate (2 accs). ----
    float hnew[2][8];
    for (int nn = 0; nn < 2; ++nn) {
      const int n0 = wave * 32 + nn * 16;
      const int mb = (lane >> 4) * 8;
      const int col = n0 + (lane & 15);

      // pass A: reset (r) and update (u) gates
      float rg[8], ug[8];
      {
        v8f a_ir = {}, a_hr = {}, a_iz = {}, a_hz = {};
#pragma unroll 2
        for (int kt = 0; kt < 16; ++kt) {
          const int kb = kt * 32;
          v16bf ax = ld_afrag(xbf, 512, kb, lane);
          v16bf ah = ld_afrag(hbf, 544, kb, lane);
          a_ir = wmma_bf16(ax, ld_bfrag(Wih, 512, n0,       kb, lane), a_ir);
          a_hr = wmma_bf16(ah, ld_bfrag(Whh, 512, n0,       kb, lane), a_hr);
          a_iz = wmma_bf16(ax, ld_bfrag(Wih, 512, n0 + 512, kb, lane), a_iz);
          a_hz = wmma_bf16(ah, ld_bfrag(Whh, 512, n0 + 512, kb, lane), a_hz);
        }
        const float bir = b_ih[col],       bhr = b_hh[col];
        const float biz = b_ih[col + 512], bhz = b_hh[col + 512];
#pragma unroll
        for (int v = 0; v < 8; ++v) {
          rg[v] = sigmoidf_(a_ir[v] + bir + a_hr[v] + bhr);
          ug[v] = sigmoidf_(a_iz[v] + biz + a_hz[v] + bhz);
        }
      }

      // pass B: candidate (n) gate + carry blend
      {
        v8f a_in = {}, a_hn = {};
#pragma unroll 2
        for (int kt = 0; kt < 16; ++kt) {
          const int kb = kt * 32;
          v16bf ax = ld_afrag(xbf, 512, kb, lane);
          v16bf ah = ld_afrag(hbf, 544, kb, lane);
          a_in = wmma_bf16(ax, ld_bfrag(Wih, 512, n0 + 1024, kb, lane), a_in);
          a_hn = wmma_bf16(ah, ld_bfrag(Whh, 512, n0 + 1024, kb, lane), a_hn);
        }
        const float bin_ = b_ih[col + 1024], bhn = b_hh[col + 1024];
#pragma unroll
        for (int v = 0; v < 8; ++v) {
          float ng = tanhf(a_in[v] + bin_ + rg[v] * (a_hn[v] + bhn));
          float ho = h_s[(mb + v) * 512 + col];
          hnew[nn][v] = (1.f - ug[v]) * ng + ug[v] * ho;
        }
      }
    }
    __syncthreads();   // all reads of old h/hbf done before overwrite
    for (int nn = 0; nn < 2; ++nn) {
      const int n0 = wave * 32 + nn * 16;
      const int mb = (lane >> 4) * 8;
      const int col = n0 + (lane & 15);
#pragma unroll
      for (int v = 0; v < 8; ++v) {
        const int m = mb + v;
        float hv = hnew[nn][v];
        h_s[m * 512 + col] = hv;
        hbf[m * 544 + col] = f2bf(hv);
        out_h[((size_t)(b0 + m) * TLEN + t) * H_DIM + col] = hv;
      }
    }
    __syncthreads();

    // ---- stage3: ha = relu([h,a] @ Wha^T + b_ha)   (K=544, N=512) ----
    for (int nn = 0; nn < 2; ++nn) {
      const int n0 = wave * 32 + nn * 16;
      v8f acc = {};
#pragma unroll 2
      for (int kt = 0; kt < 17; ++kt) {
        v16bf af = ld_afrag(hbf, 544, kt * 32, lane);
        acc = wmma_bf16(af, ld_bfrag(Wha, 544, n0, kt * 32, lane), acc);
      }
      const int mb = (lane >> 4) * 8;
      const int col = n0 + (lane & 15);
      const float bb = b_ha[col];
#pragma unroll
      for (int v = 0; v < 8; ++v)
        habf[(mb + v) * 512 + col] = f2bf(fmaxf(acc[v] + bb, 0.f));
    }
    __syncthreads();

    // ---- stage4: prior head (waves 0..3) ----
    if (wave < 4) {
      const int n0 = wave * 16;
      v8f acc = {};
#pragma unroll 2
      for (int kt = 0; kt < 16; ++kt) {
        v16bf af = ld_afrag(habf, 512, kt * 32, lane);
        acc = wmma_bf16(af, ld_bfrag(Wpr, 512, n0, kt * 32, lane), acc);
      }
      const int mb = (lane >> 4) * 8;
      const int col = n0 + (lane & 15);
      const float bb = b_prior[col];
#pragma unroll
      for (int v = 0; v < 8; ++v) {
        const int m = mb + v;
        const size_t o = ((size_t)(b0 + m) * TLEN + t) * Z_DIM;
        float x = acc[v] + bb;
        if (col < 32) out_pm[o + col] = x;
        else          out_ps[o + col - 32] = softplusf_(x);
      }
    }

    // ---- stage5: he = relu([h,e] @ Whe^T + b_he)   (K=1536, N=512) ----
    for (int nn = 0; nn < 2; ++nn) {
      const int n0 = wave * 32 + nn * 16;
      v8f acc = {};
#pragma unroll 2
      for (int kt = 0; kt < 48; ++kt) {
        const int kb = kt * 32;
        v16bf af = (kt < 16) ? ld_afrag(hbf, 544, kb, lane)
                             : ld_afrag(ebf, 1024, kb - 512, lane);
        acc = wmma_bf16(af, ld_bfrag(Whe, 1536, n0, kb, lane), acc);
      }
      const int mb = (lane >> 4) * 8;
      const int col = n0 + (lane & 15);
      const float bb = b_he[col];
#pragma unroll
      for (int v = 0; v < 8; ++v)
        hebf[(mb + v) * 512 + col] = f2bf(fmaxf(acc[v] + bb, 0.f));
    }
    __syncthreads();

    // ---- stage6: posterior head (waves 0..3) ----
    if (wave < 4) {
      const int n0 = wave * 16;
      v8f acc = {};
#pragma unroll 2
      for (int kt = 0; kt < 16; ++kt) {
        v16bf af = ld_afrag(hebf, 512, kt * 32, lane);
        acc = wmma_bf16(af, ld_bfrag(Wpo, 512, n0, kt * 32, lane), acc);
      }
      const int mb = (lane >> 4) * 8;
      const int col = n0 + (lane & 15);
      const float bb = b_post[col];
#pragma unroll
      for (int v = 0; v < 8; ++v) {
        const int m = mb + v;
        const size_t o = ((size_t)(b0 + m) * TLEN + t) * Z_DIM;
        float x = acc[v] + bb;
        if (col < 32) { qm_s[m * 32 + col] = x;      out_qm[o + col] = x; }
        else { float s = softplusf_(x);
               qs_s[m * 32 + col - 32] = s;          out_qs[o + col - 32] = s; }
      }
    }
    __syncthreads();

    // ---- stage7: z = qm + qs * eps_q (rsample), carry to next step ----
    {
      int r = tid >> 5, c = lane;
      float zn = qm_s[r * 32 + c] + qs_s[r * 32 + c] * eq_s[r * 32 + c];
      z_s[r * 32 + c] = zn;
      out_z[((size_t)(b0 + r) * TLEN + t) * Z_DIM + c] = zn;
    }
    __syncthreads();
  }
}

// ---------------------------------------------------------------------------
extern "C" void kernel_launch(void* const* d_in, const int* in_sizes, int n_in,
                              void* d_out, int out_size, void* d_ws, size_t ws_size,
                              hipStream_t stream) {
  (void)in_sizes; (void)n_in; (void)out_size; (void)ws_size;

  const float* prev_z     = (const float*)d_in[0];
  const float* prev_h     = (const float*)d_in[1];
  const float* actions    = (const float*)d_in[2];
  const float* embeddings = (const float*)d_in[3];
  const float* dones      = (const float*)d_in[4];
  // d_in[5] = prior_noise: unused by the reference outputs
  const float* post_noise = (const float*)d_in[6];
  const float* W_az    = (const float*)d_in[7];
  const float* b_az    = (const float*)d_in[8];
  const float* W_ih    = (const float*)d_in[9];
  const float* W_hh    = (const float*)d_in[10];
  const float* b_ih    = (const float*)d_in[11];
  const float* b_hh    = (const float*)d_in[12];
  const float* W_ha    = (const float*)d_in[13];
  const float* b_ha    = (const float*)d_in[14];
  const float* W_prior = (const float*)d_in[15];
  const float* b_prior = (const float*)d_in[16];
  const float* W_he    = (const float*)d_in[17];
  const float* b_he    = (const float*)d_in[18];
  const float* W_post  = (const float*)d_in[19];
  const float* b_post  = (const float*)d_in[20];

  __bf16* wbf = (__bf16*)d_ws;

  auto pack = [&](const float* src, int off, int rows, int sc, int dc) {
    int n = rows * dc;
    pack_bf16<<<(n + 255) / 256, 256, 0, stream>>>(src, wbf + off, rows, sc, dc);
  };
  pack(W_az,    OFF_WAZ,  512,  38,   64);
  pack(W_ih,    OFF_WIH, 1536, 512,  512);
  pack(W_hh,    OFF_WHH, 1536, 512,  512);
  pack(W_ha,    OFF_WHA,  512, 518,  544);
  pack(W_prior, OFF_WPR,   64, 512,  512);
  pack(W_he,    OFF_WHE,  512, 1536, 1536);
  pack(W_post,  OFF_WPO,   64, 512,  512);

  // dynamic LDS: fp32 region + bf16 region (~139 KB; 320 KB available/WGP)
  size_t smem = (size_t)(16 * 512 + 4 * 16 * 32) * sizeof(float) +
                (size_t)(16 * 544 + 16 * 1024 + 16 * 512 + 16 * 64 +
                         16 * 512 + 16 * 512) * sizeof(__bf16);

  rssm_core<<<16, 512, smem, stream>>>(prev_z, prev_h, actions, embeddings,
                                       dones, post_noise, wbf,
                                       b_az, b_ih, b_hh, b_ha, b_prior,
                                       b_he, b_post, (float*)d_out);
}
